// Interaction_25623774888013
// MI455X (gfx1250) — matrix operand order repaired and checked by reading, not compiled, not run on hardware
//
#include <hip/hip_runtime.h>
#include <hip/hip_bf16.h>

// ---------------------------------------------------------------------------
// CDNA5 (gfx1250) WMMA implementation of a SchNet CFConv + Interaction block.
//   hv  = X @ Wn + bn                           (V,64)
//   he  = ssp(ssp(Ef @ We1 + be1) @ We2 + be2)  fused per 128-edge tile in LDS
//   h   = segment_sum(hv[src] * he, dst)        (V,64) global f32 atomics (L2)
//   out = ssp(h @ Wo + bo) @ Wp + bp            (V,64)
// All GEMMs run on v_wmma_f32_16x16x32_f16 (f16 in, f32 accumulate).
// Node kernels stage C-tiles in LDS (union-aliased) for coalesced b128 stores.
// Edge kernel is grid-stride so weights are loaded to LDS once per block.
// ---------------------------------------------------------------------------

typedef _Float16 v16h __attribute__((ext_vector_type(16)));
typedef _Float16 v8h  __attribute__((ext_vector_type(8)));
typedef float    v8f  __attribute__((ext_vector_type(8)));

#define DH 64          // feature dim (D == H == 64)
#define ROWS 128       // rows per tile (8 waves x 16)
#define THREADS 256    // 8 wave32
#define EDGE_BLOCKS 1536

__device__ __forceinline__ float sspf(float x) {
  // shifted softplus: log1p(exp(x)) - log(2), numerically stable
  return fmaxf(x, 0.0f) + log1pf(__expf(-fabsf(x))) - 0.69314718055994531f;
}

// One wave computes a 16x64 tile:  C = A(16x64,f16 in LDS) * W(64x64) + bias.
// sWT is the 64x64 weight stored TRANSPOSED in LDS: sWT[n*64 + k] = W[k][n],
// so each lane's B fragment is a contiguous 32-byte LDS read.
// A fragment k order used here (16-bit A 16x32, contiguous; B uses the same):
//   lanes 0-15:  halves 0..15 -> K=kb+0..15
//   lanes 16-31: halves 0..15 -> K=kb+16..31
// B layout (32x16): halves i -> K = kb + i + (lane>=16 ? 16 : 0), N = lane&15.
__device__ __forceinline__ void wave_gemm_16x64(
    const _Float16* __restrict__ sA, int rowbase,
    const _Float16* __restrict__ sWT,
    const float* __restrict__ sbias,
    int lane, v8f acc[4])
{
  const int rA   = rowbase + (lane & 15);
  const int aoff = (lane & 16) ? 16 : 0;
  const int boff = (lane & 16) ? 16 : 0;
  const int n0   = lane & 15;
#pragma unroll
  for (int nt = 0; nt < 4; ++nt) {
    const float b = sbias[nt * 16 + n0];
    v8f c = {b, b, b, b, b, b, b, b};
#pragma unroll
    for (int kb = 0; kb < DH; kb += 32) {
      const _Float16* pa = sA + rA * DH + kb + aoff;
      v8h lo = *(const v8h*)(pa);
      v8h hi = *(const v8h*)(pa + 8);
      v16h afrag = __builtin_shufflevector(lo, hi, 0, 1, 2, 3, 4, 5, 6, 7,
                                                  8, 9, 10, 11, 12, 13, 14, 15);
      v16h bfrag = *(const v16h*)(sWT + (nt * 16 + n0) * DH + kb + boff);
      c = __builtin_amdgcn_wmma_f32_16x16x32_f16(
          /*neg_a=*/false, afrag, /*neg_b=*/false, bfrag,
          /*c_mod=*/(short)0, c, /*reuse_a=*/false, /*reuse_b=*/false);
    }
    acc[nt] = c;
  }
}

// Cooperative: load 64x64 f32 weights from global, store transposed f16 in LDS.
__device__ __forceinline__ void load_wt(const float* __restrict__ W,
                                        _Float16* __restrict__ sWT, int t) {
  for (int i = t; i < DH * DH; i += THREADS) {
    int k = i >> 6, n = i & 63;
    sWT[n * DH + k] = (_Float16)W[i];
  }
}

// ---------------------------------------------------------------------------
__global__ void zero_f32(float* __restrict__ p, int n) {
  int i = blockIdx.x * blockDim.x + threadIdx.x;
  if (i < n) p[i] = 0.0f;
}

// ---------------------------------------------------------------------------
// hv = X @ Wn + bn   (no activation), 128 rows per block, coalesced f32 out.
union HvShm {
  _Float16 x[ROWS * DH];        // input staging (f16) -- live during GEMM
  float    out[ROWS * DH];      // output staging (f32) -- live after GEMM
};

__global__ __launch_bounds__(THREADS)
void node_hv_kernel(const float* __restrict__ X,
                    const float* __restrict__ Wn, const float* __restrict__ bn,
                    float* __restrict__ hv, int V) {
  __shared__ __align__(64) HvShm    u;
  __shared__ __align__(64) _Float16 sW[DH * DH];
  __shared__ __align__(64) float    sb[DH];

  const int t = threadIdx.x;
  const int base = blockIdx.x * ROWS;
  const int nrows = min(ROWS, V - base);

  load_wt(Wn, sW, t);
  if (t < DH) sb[t] = bn[t];
  for (int i = t; i < ROWS * DH; i += THREADS)
    u.x[i] = (i < nrows * DH) ? (_Float16)X[(size_t)base * DH + i] : (_Float16)0.0f;
  __syncthreads();

  const int wave = t >> 5, lane = t & 31;
  const int rowbase = wave * 16;
  v8f acc[4];
  wave_gemm_16x64(u.x, rowbase, sW, sb, lane, acc);
  __syncthreads();                       // all reads of u.x done; reuse as f32 out

  const int n0 = lane & 15;
  const int moff = (lane & 16) ? 8 : 0;
#pragma unroll
  for (int nt = 0; nt < 4; ++nt)
#pragma unroll
    for (int r = 0; r < 8; ++r)
      u.out[(rowbase + moff + r) * DH + nt * 16 + n0] = acc[nt][r];
  __syncthreads();

  float4* dst4 = (float4*)(hv + (size_t)base * DH);
  const float4* src4 = (const float4*)u.out;
  const int n4 = nrows * (DH / 4);
  for (int i = t; i < n4; i += THREADS) dst4[i] = src4[i];
}

// ---------------------------------------------------------------------------
// Fused edge MLP + gather-multiply + scatter-add. Grid-stride over 128-edge
// tiles; weights staged in LDS once per block.
__global__ __launch_bounds__(THREADS)
void cfconv_edge_kernel(const float* __restrict__ edge_feats,
                        const float* __restrict__ We1, const float* __restrict__ be1,
                        const float* __restrict__ We2, const float* __restrict__ be2,
                        const float* __restrict__ hv,
                        const int* __restrict__ src, const int* __restrict__ dst,
                        float* __restrict__ hacc, int E) {
  __shared__ __align__(64) _Float16 sX[ROWS * DH];
  __shared__ __align__(64) _Float16 sY[ROWS * DH];
  __shared__ __align__(64) _Float16 sW1[DH * DH];
  __shared__ __align__(64) _Float16 sW2[DH * DH];
  __shared__ __align__(64) float    sb1[DH];
  __shared__ __align__(64) float    sb2[DH];
  __shared__ __align__(64) int      sSrc[ROWS];
  __shared__ __align__(64) int      sDst[ROWS];

  const int t = threadIdx.x;
  const int wave = t >> 5, lane = t & 31;
  const int rowbase = wave * 16;
  const int n0 = lane & 15;
  const int moff = (lane & 16) ? 8 : 0;
  const int numTiles = (E + ROWS - 1) / ROWS;

  load_wt(We1, sW1, t);
  load_wt(We2, sW2, t);
  if (t < DH) { sb1[t] = be1[t]; sb2[t] = be2[t]; }

  for (int tile = blockIdx.x; tile < numTiles; tile += gridDim.x) {
    const int ebase = tile * ROWS;
    const int nrows = min(ROWS, E - ebase);
    const bool full = (nrows == ROWS);

    if (t < ROWS) {
      int ok = (t < nrows);
      sSrc[t] = ok ? src[ebase + t] : 0;
      sDst[t] = ok ? dst[ebase + t] : 0;
    }
    if (full) {
      for (int i = t; i < ROWS * DH; i += THREADS)
        sX[i] = (_Float16)edge_feats[(size_t)ebase * DH + i];
    } else {
      for (int i = t; i < ROWS * DH; i += THREADS)
        sX[i] = (i < nrows * DH) ? (_Float16)edge_feats[(size_t)ebase * DH + i]
                                 : (_Float16)0.0f;
    }
    __syncthreads();

    v8f acc[4];
    // layer 1: ssp(X @ We1 + be1) -> sY (f16)
    wave_gemm_16x64(sX, rowbase, sW1, sb1, lane, acc);
#pragma unroll
    for (int nt = 0; nt < 4; ++nt)
#pragma unroll
      for (int r = 0; r < 8; ++r)
        sY[(rowbase + moff + r) * DH + nt * 16 + n0] = (_Float16)sspf(acc[nt][r]);
    __syncthreads();

    // layer 2: he = ssp(sY@We2+be2); msg = hv[src]*he; atomic scatter to hacc
    wave_gemm_16x64(sY, rowbase, sW2, sb2, lane, acc);
    if (full) {
#pragma unroll
      for (int nt = 0; nt < 4; ++nt) {
        const int f = nt * 16 + n0;
#pragma unroll
        for (int r = 0; r < 8; ++r) {
          int row = rowbase + moff + r;
          float he = sspf(acc[nt][r]);
          float m = he * hv[(size_t)sSrc[row] * DH + f];
          atomicAdd(&hacc[(size_t)sDst[row] * DH + f], m);
        }
      }
    } else {
#pragma unroll
      for (int nt = 0; nt < 4; ++nt) {
        const int f = nt * 16 + n0;
#pragma unroll
        for (int r = 0; r < 8; ++r) {
          int row = rowbase + moff + r;
          if (row < nrows) {
            float he = sspf(acc[nt][r]);
            float m = he * hv[(size_t)sSrc[row] * DH + f];
            atomicAdd(&hacc[(size_t)sDst[row] * DH + f], m);
          }
        }
      }
    }
    __syncthreads();   // protect sX/sY/sSrc/sDst before next tile overwrites
  }
}

// ---------------------------------------------------------------------------
// out = ssp(h @ Wo + bo) @ Wp + bp  (two chained WMMA GEMMs), coalesced out.
union OutShm {
  struct { _Float16 x[ROWS * DH]; _Float16 y[ROWS * DH]; } s;  // GEMM staging
  float out[ROWS * DH];                                        // f32 out staging
};

__global__ __launch_bounds__(THREADS)
void node_out_kernel(const float* __restrict__ Hin,
                     const float* __restrict__ Wo, const float* __restrict__ bo,
                     const float* __restrict__ Wp, const float* __restrict__ bp,
                     float* __restrict__ out, int V) {
  __shared__ __align__(64) OutShm   u;
  __shared__ __align__(64) _Float16 sWo[DH * DH];
  __shared__ __align__(64) _Float16 sWp[DH * DH];
  __shared__ __align__(64) float    sbo[DH];
  __shared__ __align__(64) float    sbp[DH];

  const int t = threadIdx.x;
  const int base = blockIdx.x * ROWS;
  const int nrows = min(ROWS, V - base);

  load_wt(Wo, sWo, t);
  load_wt(Wp, sWp, t);
  if (t < DH) { sbo[t] = bo[t]; sbp[t] = bp[t]; }
  for (int i = t; i < ROWS * DH; i += THREADS)
    u.s.x[i] = (i < nrows * DH) ? (_Float16)Hin[(size_t)base * DH + i]
                                : (_Float16)0.0f;
  __syncthreads();

  const int wave = t >> 5, lane = t & 31;
  const int rowbase = wave * 16;
  const int n0 = lane & 15;
  const int moff = (lane & 16) ? 8 : 0;

  v8f acc[4];
  wave_gemm_16x64(u.s.x, rowbase, sWo, sbo, lane, acc);
#pragma unroll
  for (int nt = 0; nt < 4; ++nt)
#pragma unroll
    for (int r = 0; r < 8; ++r)
      u.s.y[(rowbase + moff + r) * DH + nt * 16 + n0] = (_Float16)sspf(acc[nt][r]);
  __syncthreads();

  wave_gemm_16x64(u.s.y, rowbase, sWp, sbp, lane, acc);
  __syncthreads();                 // all reads of u.s.* done; reuse as f32 out
#pragma unroll
  for (int nt = 0; nt < 4; ++nt)
#pragma unroll
    for (int r = 0; r < 8; ++r)
      u.out[(rowbase + moff + r) * DH + nt * 16 + n0] = acc[nt][r];
  __syncthreads();

  float4* dst4 = (float4*)(out + (size_t)base * DH);
  const float4* src4 = (const float4*)u.out;
  const int n4 = nrows * (DH / 4);
  for (int i = t; i < n4; i += THREADS) dst4[i] = src4[i];
}

// ---------------------------------------------------------------------------
extern "C" void kernel_launch(void* const* d_in, const int* in_sizes, int n_in,
                              void* d_out, int out_size, void* d_ws, size_t ws_size,
                              hipStream_t stream) {
  const float* node_feats = (const float*)d_in[0];
  const float* edge_feats = (const float*)d_in[1];
  const int*   src        = (const int*)d_in[2];
  const int*   dst        = (const int*)d_in[3];
  const float* Wn  = (const float*)d_in[4];
  const float* bn  = (const float*)d_in[5];
  const float* We1 = (const float*)d_in[6];
  const float* be1 = (const float*)d_in[7];
  const float* We2 = (const float*)d_in[8];
  const float* be2 = (const float*)d_in[9];
  const float* Wo  = (const float*)d_in[10];
  const float* bo  = (const float*)d_in[11];
  const float* Wp  = (const float*)d_in[12];
  const float* bp  = (const float*)d_in[13];

  const int V = in_sizes[0] / DH;
  const int E = in_sizes[2];

  float* hv   = (float*)d_ws;                  // V*64 f32
  float* hacc = hv + (size_t)V * DH;           // V*64 f32

  const int nAcc = V * DH;
  zero_f32<<<(nAcc + 255) / 256, 256, 0, stream>>>(hacc, nAcc);

  const int nodeBlocks = (V + ROWS - 1) / ROWS;
  node_hv_kernel<<<nodeBlocks, THREADS, 0, stream>>>(node_feats, Wn, bn, hv, V);

  const int edgeTiles = (E + ROWS - 1) / ROWS;
  const int edgeBlocks = edgeTiles < EDGE_BLOCKS ? edgeTiles : EDGE_BLOCKS;
  cfconv_edge_kernel<<<edgeBlocks, THREADS, 0, stream>>>(
      edge_feats, We1, be1, We2, be2, hv, src, dst, hacc, E);

  node_out_kernel<<<nodeBlocks, THREADS, 0, stream>>>(
      hacc, Wo, bo, Wp, bp, (float*)d_out, V);
}
